// Decoder_57655640981685
// MI455X (gfx1250) — matrix operand order repaired and checked by reading, not compiled, not run on hardware
//
#include <hip/hip_runtime.h>
#include <hip/hip_bf16.h>
#include <stdint.h>

// ---------------------------------------------------------------------------
// LAS-style attention LSTM decoder for MI455X (gfx1250, wave32, WMMA).
//
// Roofline: attention reads keys+values each of the 500 steps. In f32 that is
// 256 MB/step -> 128 GB total (5.5 ms at 23.3 TB/s HBM). Converted once to
// bf16 the pair is 128 MB, which fits the 192 MB L2, so the steady-state
// stream comes from L2. All GEMMs (gates/proj/query) use
// v_wmma_f32_16x16x32_bf16 with f32 accumulation; softmax and LSTM cell state
// stay f32. Deterministic: chunked softmax combined in fixed order, no float
// atomics. 5 kernels per step (fused LSTM+query, fused proj+argmax, fused
// softmax-combine+emb-gather) to minimize graph-node dispatch latency on this
// 500-long dependency chain.
// ---------------------------------------------------------------------------

#define E_DIM   400
#define H_DIM   300
#define K_DIM   128
#define V_DIM   128
#define VOCAB_N 33
#define MAXLEN  500
#define SOS_TOK 5
#define T_LEN   2048
#define B_SZ    128

#define XH_LD   832   // 528 (emb+ctx) + 300 (h) + 4 pad  -> 26 k-tiles of 32
#define HCTX_LD 448   // 300 (h) + 128 (ctx) + 20 pad     -> 14 k-tiles
#define HPAD_LD 320   // 300 (h) + 20 pad                 -> 10 k-tiles
#define G_LD    1200
#define CH      64
#define NCH     (T_LEN / CH)

typedef __bf16 bf16_t;
typedef __attribute__((ext_vector_type(16))) __bf16 v16bf;
typedef __attribute__((ext_vector_type(8)))  __bf16 v8bf;
typedef __attribute__((ext_vector_type(8)))  float  v8f;

__device__ __forceinline__ float sigf(float x) { return 1.0f / (1.0f + __expf(-x)); }

// WMMA helper: one 16x16x32 bf16 MACC given per-lane fragment pointers.
// Fragment layouts used by this code (contiguous k, not the ISA table order):
//   A 16x32 bf16 : lane L holds row (L&15); K range (L>>4)*16..+15
//                                                            (two 16B loads)
//   B 32x16 bf16 : lane L holds col (L&15); K range (L>>4)*16..+15 (32B load)
//   C/D 16x16 f32: lane L col (L&15); VGPR v -> row v + 8*(L>>4)
__device__ __forceinline__ v8f wmma_step(const bf16_t* ap, const bf16_t* bp, v8f acc) {
  v16bf bfrag = *(const v16bf*)bp;
  v8bf lo = *(const v8bf*)(ap);
  v8bf hi = *(const v8bf*)(ap + 8);
  v16bf afrag;
#pragma unroll
  for (int u = 0; u < 8; ++u) { afrag[u] = lo[u]; afrag[u + 8] = hi[u]; }
  return __builtin_amdgcn_wmma_f32_16x16x32_bf16(false, afrag, false, bfrag,
                                                 (short)0, acc, false, false);
}

// ---------------------------------------------------------------------------
// Generic bf16 WMMA GEMM:  C[M=128, N] = A[M, K] * B[N, K]^T + bias[N]
// One wave per block. grid.x = N/16 tiles, grid.y = 2 (rows 0-63 / 64-127).
// Used for the gates GEMM (the big one) and the init-time query GEMM.
// ---------------------------------------------------------------------------
__global__ void wmma_gemm_bf16(const bf16_t* __restrict__ A, int lda,
                               const bf16_t* __restrict__ B, int ldb,
                               const float* __restrict__ bias,
                               float* __restrict__ C, int ldc,
                               int kTiles) {
  const int nt   = blockIdx.x;
  const int mq   = blockIdx.y;
  const int lane = threadIdx.x;         // 32 threads = 1 wave
  const int lrow = lane & 15;
  const int lhi  = lane >> 4;

  v8f acc[4] = {};
  const bf16_t* Arow = A + (size_t)(mq * 64 + lrow) * lda;
  const bf16_t* Brow = B + (size_t)(nt * 16 + lrow) * ldb + lhi * 16;

  for (int kt = 0; kt < kTiles; ++kt) {
    const int kk = kt * 32;
#pragma unroll
    for (int m = 0; m < 4; ++m)
      acc[m] = wmma_step(Arow + (size_t)m * 16 * lda + kk + lhi * 16, Brow + kk, acc[m]);
  }

  const int nCol = nt * 16 + lrow;
  const float bs = bias ? bias[nCol] : 0.0f;
#pragma unroll
  for (int m = 0; m < 4; ++m) {
#pragma unroll
    for (int v = 0; v < 8; ++v) {
      const int row = mq * 64 + m * 16 + lhi * 8 + v;
      C[(size_t)row * ldc + nCol] = acc[m][v] + bs;
    }
  }
}

// ---------------------------------------------------------------------------
// One-time conversion / packing kernels
// ---------------------------------------------------------------------------
__global__ void cvt_f32_bf16(const float* __restrict__ src, bf16_t* __restrict__ dst,
                             size_t n) {
  const size_t stride = (size_t)gridDim.x * blockDim.x;
  for (size_t i = (size_t)blockIdx.x * blockDim.x + threadIdx.x; i < n; i += stride)
    dst[i] = (bf16_t)src[i];
}

// W_cat[1200, 832] = [W_ih (528 cols) | W_hh (300 cols) | zero pad]
__global__ void pack_wcat(const float* __restrict__ Wih, const float* __restrict__ Whh,
                          bf16_t* __restrict__ Wcat) {
  const int total = G_LD * XH_LD;
  const int stride = gridDim.x * blockDim.x;
  for (int i = blockIdx.x * blockDim.x + threadIdx.x; i < total; i += stride) {
    const int r = i / XH_LD, k = i - r * XH_LD;
    float v;
    if (k < 528)      v = Wih[r * 528 + k];
    else if (k < 828) v = Whh[r * 300 + (k - 528)];
    else              v = 0.0f;
    Wcat[i] = (bf16_t)v;
  }
}

__global__ void pack_pad(const float* __restrict__ src, bf16_t* __restrict__ dst,
                         int rows, int cols, int srows, int scols) {
  const int total = rows * cols;
  const int stride = gridDim.x * blockDim.x;
  for (int i = blockIdx.x * blockDim.x + threadIdx.x; i < total; i += stride) {
    const int r = i / cols, c = i - r * cols;
    const float v = (r < srows && c < scols) ? src[r * scols + c] : 0.0f;
    dst[i] = (bf16_t)v;
  }
}

__global__ void add_bias2(const float* __restrict__ a, const float* __restrict__ b,
                          float* __restrict__ out, int n) {
  const int i = blockIdx.x * blockDim.x + threadIdx.x;
  if (i < n) out[i] = a[i] + b[i];
}

__global__ void pad_bias(const float* __restrict__ src, float* __restrict__ dst,
                         int n, int npad) {
  const int i = blockIdx.x * blockDim.x + threadIdx.x;
  if (i < npad) dst[i] = (i < n) ? src[i] : 0.0f;
}

// h parts <- h0 broadcast, pads <- 0, c <- c0 broadcast, char <- SOS
__global__ void init_state(bf16_t* __restrict__ xh, bf16_t* __restrict__ hctx,
                           bf16_t* __restrict__ hpad, float* __restrict__ c,
                           int* __restrict__ charbuf,
                           const float* __restrict__ h0, const float* __restrict__ c0) {
  const int stride = gridDim.x * blockDim.x;
  const int tid = blockIdx.x * blockDim.x + threadIdx.x;
  for (int i = tid; i < B_SZ * XH_LD; i += stride) {
    const int col = i % XH_LD;
    xh[i] = (bf16_t)((col >= 528 && col < 828) ? h0[col - 528] : 0.0f);
  }
  for (int i = tid; i < B_SZ * HCTX_LD; i += stride) {
    const int col = i % HCTX_LD;
    hctx[i] = (bf16_t)(col < H_DIM ? h0[col] : 0.0f);
  }
  for (int i = tid; i < B_SZ * HPAD_LD; i += stride) {
    const int col = i % HPAD_LD;
    hpad[i] = (bf16_t)(col < H_DIM ? h0[col] : 0.0f);
  }
  for (int i = tid; i < B_SZ * H_DIM; i += stride) c[i] = c0[i % H_DIM];
  for (int i = tid; i < B_SZ; i += stride) charbuf[i] = SOS_TOK;
}

// ---------------------------------------------------------------------------
// Fused LSTM elementwise + query GEMM.
// One block, 512 threads = 16 waves. LSTM writes bf16 h to LDS (and to the
// global GEMM input buffers), then the 16 waves compute
// q[128,128] = h @ W_phi^T + b_phi with WMMA, A-fragments from LDS.
// All per-thread loops have exact trip counts -> no divergence before WMMA.
// ---------------------------------------------------------------------------
__global__ void lstm_query_fused(const float* __restrict__ gates, float* __restrict__ c,
                                 bf16_t* __restrict__ xh, bf16_t* __restrict__ hctx,
                                 const bf16_t* __restrict__ Wphi,
                                 const float* __restrict__ bphi,
                                 float* __restrict__ q) {
  __shared__ bf16_t hs[B_SZ][HPAD_LD];            // 80 KB of 320 KB LDS
  const int tid = threadIdx.x;                    // 512
  for (int i = tid; i < B_SZ * (HPAD_LD - H_DIM); i += 512) {  // zero pad cols
    const int b = i / (HPAD_LD - H_DIM), j = i % (HPAD_LD - H_DIM);
    hs[b][H_DIM + j] = (bf16_t)0.0f;
  }
  for (int i = tid; i < B_SZ * H_DIM; i += 512) { // 75 exact iters/thread
    const int b = i / H_DIM, j = i - b * H_DIM;
    const float* g = gates + (size_t)b * G_LD;
    const float ig = sigf(g[j]);
    const float fg = sigf(g[j + H_DIM]);
    const float gg = tanhf(g[j + 2 * H_DIM]);
    const float og = sigf(g[j + 3 * H_DIM]);
    const float cn = fg * c[i] + ig * gg;
    const float hn = og * tanhf(cn);
    c[i] = cn;
    const bf16_t hb = (bf16_t)hn;
    hs[b][j] = hb;
    xh[(size_t)b * XH_LD + 528 + j] = hb;
    hctx[(size_t)b * HCTX_LD + j]   = hb;
  }
  __syncthreads();

  const int wave = tid >> 5, lane = tid & 31;
  const int nt = wave & 7, mq = wave >> 3;
  const int lrow = lane & 15, lhi = lane >> 4;
  v8f acc[4] = {};
  const bf16_t* Brow = Wphi + (size_t)(nt * 16 + lrow) * HPAD_LD + lhi * 16;
  for (int kt = 0; kt < HPAD_LD / 32; ++kt) {
    const int kk = kt * 32;
#pragma unroll
    for (int m = 0; m < 4; ++m)
      acc[m] = wmma_step(&hs[mq * 64 + m * 16 + lrow][kk + lhi * 16], Brow + kk, acc[m]);
  }
  const int nCol = nt * 16 + lrow;
  const float bs = bphi[nCol];
#pragma unroll
  for (int m = 0; m < 4; ++m) {
#pragma unroll
    for (int v = 0; v < 8; ++v)
      q[(size_t)(mq * 64 + m * 16 + lhi * 8 + v) * K_DIM + nCol] = acc[m][v] + bs;
  }
}

// ---------------------------------------------------------------------------
// Fused projection GEMM + logits writeback + argmax.
// One block, 192 threads = 6 waves covering (nt 0..2) x (mq 0..1); logits
// staged in LDS, then per-row argmax (first-max == jnp.argmax semantics).
// ---------------------------------------------------------------------------
__global__ void proj_argmax_fused(const bf16_t* __restrict__ A /*hctx*/,
                                  const bf16_t* __restrict__ B /*Wproj padded*/,
                                  const float* __restrict__ bias,
                                  float* __restrict__ out_logits,
                                  int* __restrict__ out_preds,
                                  int* __restrict__ charbuf, int step) {
  __shared__ float ls[B_SZ][48];                  // 24 KB
  const int tid = threadIdx.x;                    // 192
  const int wave = tid >> 5, lane = tid & 31;
  const int nt = wave % 3, mq = wave / 3;
  const int lrow = lane & 15, lhi = lane >> 4;
  v8f acc[4] = {};
  const bf16_t* Arow = A + (size_t)(mq * 64 + lrow) * HCTX_LD;
  const bf16_t* Brow = B + (size_t)(nt * 16 + lrow) * HCTX_LD + lhi * 16;
  for (int kt = 0; kt < HCTX_LD / 32; ++kt) {
    const int kk = kt * 32;
#pragma unroll
    for (int m = 0; m < 4; ++m)
      acc[m] = wmma_step(Arow + (size_t)m * 16 * HCTX_LD + kk + lhi * 16, Brow + kk, acc[m]);
  }
  const int nCol = nt * 16 + lrow;
  const float bs = bias[nCol];
#pragma unroll
  for (int m = 0; m < 4; ++m) {
#pragma unroll
    for (int v = 0; v < 8; ++v)
      ls[mq * 64 + m * 16 + lhi * 8 + v][nCol] = acc[m][v] + bs;
  }
  __syncthreads();
  if (tid < B_SZ) {
    const int b = tid;
    float* orow = out_logits + ((size_t)step * B_SZ + b) * VOCAB_N;
    float best = ls[b][0]; int bi = 0;
    orow[0] = best;
    for (int k = 1; k < VOCAB_N; ++k) {
      const float v = ls[b][k];
      orow[k] = v;
      if (v > best) { best = v; bi = k; }
    }
    out_preds[(size_t)step * B_SZ + b] = bi;
    charbuf[b] = bi;
  }
}

// ---------------------------------------------------------------------------
// Attention pass 1: per (T-chunk of 64, b-group of 16) compute energies,
// chunk-local max/sumexp, and partial weighted-value context. 32B key loads;
// global_prefetch of exactly the values rows phase C will consume.
// ---------------------------------------------------------------------------
__global__ void attn_pass1(const bf16_t* __restrict__ kbuf, const bf16_t* __restrict__ vbuf,
                           const float* __restrict__ q,
                           float* __restrict__ pm, float* __restrict__ ps,
                           float* __restrict__ pctx) {
  __shared__ float qs[16][K_DIM];
  __shared__ float esh[16][CH];
  const int chk = blockIdx.x, bg = blockIdx.y;
  const int b0 = bg * 16, t0 = chk * CH;
  const int tid = threadIdx.x;                    // 256 threads
  for (int i = tid; i < 16 * K_DIM; i += 256)
    qs[i >> 7][i & 127] = q[(size_t)(b0 + (i >> 7)) * K_DIM + (i & 127)];
  __syncthreads();

  const int bl  = tid & 15;
  const int sub = tid >> 4;                       // 0..15
#pragma unroll
  for (int j = 0; j < CH / 16; ++j) {             // e[b,t] = q[b] . keys[t,b]
    const int tl = sub + 16 * j;
    const size_t row = (size_t)(t0 + tl) * B_SZ + (b0 + bl);
    __builtin_prefetch(vbuf + row * V_DIM, 0, 3); // warm values for phase C
    const bf16_t* kp = kbuf + row * K_DIM;
    float acc = 0.0f;
#pragma unroll
    for (int k = 0; k < K_DIM; k += 16) {
      v16bf kv = *(const v16bf*)(kp + k);
#pragma unroll
      for (int u = 0; u < 16; ++u) acc += qs[bl][k + u] * (float)kv[u];
    }
    esh[bl][tl] = acc;
  }
  __syncthreads();

  if (tid < 16) {                                 // chunk-local softmax stats
    float m = -1e30f;
    for (int t = 0; t < CH; ++t) m = fmaxf(m, esh[tid][t]);
    float s = 0.0f;
    for (int t = 0; t < CH; ++t) { const float p = __expf(esh[tid][t] - m); esh[tid][t] = p; s += p; }
    pm[chk * B_SZ + b0 + tid] = m;
    ps[chk * B_SZ + b0 + tid] = s;
  }
  __syncthreads();

  float acc[8];                                   // partial ctx
#pragma unroll
  for (int u = 0; u < 8; ++u) acc[u] = 0.0f;
  for (int tl = 0; tl < CH; ++tl) {
    const float w = esh[bl][tl];
    const bf16_t* vp = vbuf + ((size_t)(t0 + tl) * B_SZ + (b0 + bl)) * V_DIM + sub * 8;
    v8bf vv = *(const v8bf*)vp;
#pragma unroll
    for (int u = 0; u < 8; ++u) acc[u] += w * (float)vv[u];
  }
  float* outp = pctx + ((size_t)chk * B_SZ + b0 + bl) * V_DIM + sub * 8;
#pragma unroll
  for (int u = 0; u < 8; ++u) outp[u] = acc[u];
}

// ---------------------------------------------------------------------------
// Attention pass 2 (exact fixed-order combine over 32 chunks) fused with the
// emb-row gather for the NEXT step (charbuf is final by the time this runs).
// ---------------------------------------------------------------------------
__global__ void attn_pass2_gather(const float* __restrict__ pm, const float* __restrict__ ps,
                                  const float* __restrict__ pctx,
                                  bf16_t* __restrict__ xh, bf16_t* __restrict__ hctx,
                                  const bf16_t* __restrict__ emb,
                                  const int* __restrict__ charbuf) {
  const int b = blockIdx.x;
  const int tid = threadIdx.x;                    // 128
  __shared__ float sm[NCH], ss[NCH], w[NCH];
  __shared__ float Sinv;
  if (tid < NCH) { sm[tid] = pm[tid * B_SZ + b]; ss[tid] = ps[tid * B_SZ + b]; }
  __syncthreads();
  if (tid == 0) {
    float M = -1e30f;
    for (int i = 0; i < NCH; ++i) M = fmaxf(M, sm[i]);
    float S = 0.0f;
    for (int i = 0; i < NCH; ++i) { w[i] = __expf(sm[i] - M); S += w[i] * ss[i]; }
    Sinv = 1.0f / S;
  }
  __syncthreads();
  float acc = 0.0f;
  for (int i = 0; i < NCH; ++i)
    acc += w[i] * pctx[((size_t)i * B_SZ + b) * V_DIM + tid];
  const float ctx = acc * Sinv;
  xh[(size_t)b * XH_LD + 400 + tid]     = (bf16_t)ctx;
  hctx[(size_t)b * HCTX_LD + 300 + tid] = (bf16_t)ctx;
  // emb gather for the next step's x
  const int ch = charbuf[b];
  const bf16_t* src = emb + (size_t)ch * E_DIM;
  bf16_t* dst = xh + (size_t)b * XH_LD;
  for (int j = tid; j < E_DIM; j += 128) dst[j] = src[j];
}

// ---------------------------------------------------------------------------
// Host launcher (graph-capture safe: only kernel launches on `stream`)
// ---------------------------------------------------------------------------
extern "C" void kernel_launch(void* const* d_in, const int* in_sizes, int n_in,
                              void* d_out, int out_size, void* d_ws, size_t ws_size,
                              hipStream_t stream) {
  (void)in_sizes; (void)n_in; (void)out_size;
  const float* keys   = (const float*)d_in[0];
  const float* values = (const float*)d_in[1];
  const float* emb    = (const float*)d_in[2];
  const float* W_phi  = (const float*)d_in[3];
  const float* b_phi  = (const float*)d_in[4];
  const float* W_ih   = (const float*)d_in[5];
  const float* b_ih   = (const float*)d_in[6];
  const float* W_hh   = (const float*)d_in[7];
  const float* b_hh   = (const float*)d_in[8];
  const float* W_proj = (const float*)d_in[9];
  const float* b_proj = (const float*)d_in[10];
  const float* h0     = (const float*)d_in[11];
  const float* c0     = (const float*)d_in[12];

  char* ws = (char*)d_ws;
  size_t off = 0;
  auto take = [&](size_t bytes) {
    size_t o = off;
    off = (off + bytes + 255) & ~(size_t)255;
    return o;
  };

  const size_t NKV = (size_t)T_LEN * B_SZ * K_DIM;     // 33.5M elements
  bf16_t* kbuf    = (bf16_t*)(ws + take(NKV * 2));     // 64 MB  (L2-resident)
  bf16_t* vbuf    = (bf16_t*)(ws + take(NKV * 2));     // 64 MB
  bf16_t* Wcat    = (bf16_t*)(ws + take((size_t)G_LD * XH_LD * 2));
  float*  bcat    = (float*) (ws + take(G_LD * 4));
  bf16_t* Wphip   = (bf16_t*)(ws + take((size_t)K_DIM * HPAD_LD * 2));
  bf16_t* Wprojp  = (bf16_t*)(ws + take((size_t)48 * HCTX_LD * 2));
  float*  bprojp  = (float*) (ws + take(48 * 4));
  bf16_t* embb    = (bf16_t*)(ws + take((size_t)VOCAB_N * E_DIM * 2));
  bf16_t* xh      = (bf16_t*)(ws + take((size_t)B_SZ * XH_LD * 2));
  bf16_t* hctx    = (bf16_t*)(ws + take((size_t)B_SZ * HCTX_LD * 2));
  bf16_t* hpad    = (bf16_t*)(ws + take((size_t)B_SZ * HPAD_LD * 2)); // init only
  float*  gates   = (float*) (ws + take((size_t)B_SZ * G_LD * 4));
  float*  cbuf    = (float*) (ws + take((size_t)B_SZ * H_DIM * 4));
  float*  qbuf    = (float*) (ws + take((size_t)B_SZ * K_DIM * 4));
  int*    charbuf = (int*)   (ws + take(B_SZ * 4));
  float*  pm      = (float*) (ws + take((size_t)NCH * B_SZ * 4));
  float*  ps      = (float*) (ws + take((size_t)NCH * B_SZ * 4));
  float*  pctx    = (float*) (ws + take((size_t)NCH * B_SZ * V_DIM * 4));
  if (off > ws_size) return;

  float* outLogits = (float*)d_out;                     // [500,128,33] f32
  int*   outPreds  = (int*)((float*)d_out + (size_t)MAXLEN * B_SZ * VOCAB_N);

  // ---- one-time packing (recomputed every call: stateless/deterministic) ----
  cvt_f32_bf16<<<4096, 256, 0, stream>>>(keys,   kbuf, NKV);
  cvt_f32_bf16<<<4096, 256, 0, stream>>>(values, vbuf, NKV);
  pack_wcat   <<<1024, 256, 0, stream>>>(W_ih, W_hh, Wcat);
  add_bias2   <<<5,    256, 0, stream>>>(b_ih, b_hh, bcat, G_LD);
  pack_pad    <<<160,  256, 0, stream>>>(W_phi,  Wphip,  K_DIM, HPAD_LD, K_DIM, H_DIM);
  pack_pad    <<<84,   256, 0, stream>>>(W_proj, Wprojp, 48, HCTX_LD, VOCAB_N, H_DIM + V_DIM);
  pack_pad    <<<52,   256, 0, stream>>>(emb,    embb,   VOCAB_N, E_DIM, VOCAB_N, E_DIM);
  pad_bias    <<<1,    64,  0, stream>>>(b_proj, bprojp, VOCAB_N, 48);
  init_state  <<<256,  256, 0, stream>>>(xh, hctx, hpad, cbuf, charbuf, h0, c0);

  // ---- initial attention with h = h0 (q = h0 @ W_phi^T + b_phi), plus the
  //      SOS emb gather for step 0 ----
  wmma_gemm_bf16<<<dim3(8, 2), 32, 0, stream>>>(hpad, HPAD_LD, Wphip, HPAD_LD,
                                                b_phi, qbuf, K_DIM, HPAD_LD / 32);
  attn_pass1<<<dim3(NCH, 8), 256, 0, stream>>>(kbuf, vbuf, qbuf, pm, ps, pctx);
  attn_pass2_gather<<<B_SZ, 128, 0, stream>>>(pm, ps, pctx, xh, hctx, embb, charbuf);

  // ---- decode loop: 5 kernels per step ----
  for (int step = 0; step < MAXLEN; ++step) {
    // gates = [emb|ctx|h] @ [W_ih|W_hh]^T + (b_ih + b_hh)
    wmma_gemm_bf16<<<dim3(G_LD / 16, 2), 32, 0, stream>>>(xh, XH_LD, Wcat, XH_LD,
                                                          bcat, gates, G_LD, XH_LD / 32);
    // LSTM cell + q = h @ W_phi^T + b_phi (WMMA from LDS)
    lstm_query_fused<<<1, 512, 0, stream>>>(gates, cbuf, xh, hctx, Wphip, b_phi, qbuf);
    // logits = [h|ctx] @ W_proj^T + b_proj, argmax, writeback
    proj_argmax_fused<<<1, 192, 0, stream>>>(hctx, Wprojp, bprojp,
                                             outLogits, outPreds, charbuf, step);
    if (step + 1 < MAXLEN) {                    // context + emb gather for next step
      attn_pass1<<<dim3(NCH, 8), 256, 0, stream>>>(kbuf, vbuf, qbuf, pm, ps, pctx);
      attn_pass2_gather<<<B_SZ, 128, 0, stream>>>(pm, ps, pctx, xh, hctx, embb, charbuf);
    }
  }
}
